// NN_CRF_37125697307273
// MI455X (gfx1250) — matrix operand order repaired and checked by reading, not compiled, not run on hardware
//
#include <hip/hip_runtime.h>
#include <hip/hip_bf16.h>

typedef __attribute__((ext_vector_type(16))) _Float16 v16h;
typedef __attribute__((ext_vector_type(8)))  _Float16 v8h;
typedef __attribute__((ext_vector_type(8)))  float    v8f;

#define Bsz 512
#define Lsz 512
#define Tsz 128
#define BPG 16                    // batch rows per workgroup (WMMA M)
#define PASTRIDE (Tsz + 8)        // padded f16 row stride: +8 halves => banks rotate by 4/row

// ---------------------------------------------------------------------------
// k0: E^T[j][i] = exp(trans[i][j]) as f16, stored transposed 128x128 (WMMA-B friendly:
//     lane reads column j0+Ml, K={8h..8h+7, 16+8h..16+8h+7} of each 32-chunk)
// ---------------------------------------------------------------------------
__global__ void crf_prep_expT(const float* __restrict__ trans,
                              _Float16* __restrict__ expT) {
  int i = blockIdx.x * blockDim.x + threadIdx.x;
  if (i < Tsz * Tsz) expT[(i % Tsz) * Tsz + i / Tsz] = (_Float16)__expf(trans[i]);
}

// ---------------------------------------------------------------------------
// k1: forward recurrence. 32 WGs x 256 threads (8 wave32). Wave w owns the
//     16-column j-tile; 4x v_wmma_f32_16x16x32_f16 per step, B operand kept
//     in VGPRs across all 511 steps.
// ---------------------------------------------------------------------------
__global__ __launch_bounds__(256) void crf_forward(
    const float* __restrict__ in, const float* __restrict__ startT,
    const float* __restrict__ endT, const int* __restrict__ mask,
    const _Float16* __restrict__ expT, float* __restrict__ logden) {
  __shared__ float alphaS[BPG][Tsz];                 // running alpha (f32)
  __shared__ __align__(16) _Float16 pA[BPG][PASTRIDE]; // p = exp(alpha - m), f16
  __shared__ float emitS[BPG][Tsz];                  // staged emissions
  __shared__ float mArr[BPG];                        // per-row max
  __shared__ float mkArr[BPG];                       // per-row mask(t)

  const int tid  = threadIdx.x;
  const int b0   = blockIdx.x * BPG;
  const int r    = tid >> 4;        // phase-A row 0..15
  const int q    = tid & 15;        // phase-A col group
  const int c0   = q << 3;          // 8 columns per thread
  const int ld   = tid & 31;        // lane in wave
  const int w    = tid >> 5;        // wave id 0..7 -> j-tile
  const int Ml   = ld & 15;
  const int hi16 = ld >> 4;
  const int j0   = w << 4;

  // Preload WMMA B operand: expT tile columns [j0, j0+16), all K. 32 VGPRs.
  v16h bfrag[4];
#pragma unroll
  for (int kc = 0; kc < 4; ++kc) {
    const _Float16* bp = expT + (size_t)(j0 + Ml) * Tsz + 32 * kc + 8 * hi16;
    v8h bl = *(const v8h*)bp;
    v8h bh = *(const v8h*)(bp + 16);
    for (int i = 0; i < 8; ++i) { bfrag[kc][i] = bl[i]; bfrag[kc][i + 8] = bh[i]; }
  }

  // alpha0 = start + emit[0]
  {
    const float* ep = in + ((size_t)(b0 + r) * Lsz) * Tsz + c0;
    float4 e0 = ((const float4*)ep)[0];
    float4 e1 = ((const float4*)ep)[1];
    float4 s0 = ((const float4*)(startT + c0))[0];
    float4 s1 = ((const float4*)(startT + c0))[1];
    alphaS[r][c0 + 0] = s0.x + e0.x;  alphaS[r][c0 + 1] = s0.y + e0.y;
    alphaS[r][c0 + 2] = s0.z + e0.z;  alphaS[r][c0 + 3] = s0.w + e0.w;
    alphaS[r][c0 + 4] = s1.x + e1.x;  alphaS[r][c0 + 5] = s1.y + e1.y;
    alphaS[r][c0 + 6] = s1.z + e1.z;  alphaS[r][c0 + 7] = s1.w + e1.w;
  }
  __syncthreads();

  for (int t = 1; t < Lsz; ++t) {
    // ---- Phase A: row max, p = exp(alpha-m) -> f16 LDS, stage emit[t] ----
    float av[8];
#pragma unroll
    for (int i = 0; i < 8; ++i) av[i] = alphaS[r][c0 + i];
    float mx = av[0];
#pragma unroll
    for (int i = 1; i < 8; ++i) mx = fmaxf(mx, av[i]);
#pragma unroll
    for (int off = 1; off < 16; off <<= 1) mx = fmaxf(mx, __shfl_xor(mx, off));
    v8h p;
#pragma unroll
    for (int i = 0; i < 8; ++i) p[i] = (_Float16)__expf(av[i] - mx);
    *(v8h*)&pA[r][c0] = p;

    const float* ep = in + ((size_t)(b0 + r) * Lsz + t) * Tsz + c0;
    float4 e0 = ((const float4*)ep)[0];
    float4 e1 = ((const float4*)ep)[1];
    ((float4*)&emitS[r][c0])[0] = e0;
    ((float4*)&emitS[r][c0])[1] = e1;
    __builtin_prefetch(ep + Tsz, 0, 1);   // next timestep -> global_prefetch_b8
    if (q == 0) {
      mArr[r]  = mx;
      mkArr[r] = (float)mask[(size_t)(b0 + r) * Lsz + t];
    }
    __syncthreads();

    // ---- Phase B: acc = P(16x128) x E(128x16) via 4 chained WMMAs ----
    v8f acc = {};
#pragma unroll
    for (int kc = 0; kc < 4; ++kc) {
      // ISA 16-bit A layout: lanes 0-15 hold K={0-7,16-23}, lanes 16-31 K={8-15,24-31}
      const int ka = 32 * kc + (hi16 ? 8 : 0);
      const int kb = 32 * kc + (hi16 ? 24 : 16);
      v8h lo = *(const v8h*)&pA[Ml][ka];
      v8h hi = *(const v8h*)&pA[Ml][kb];
      v16h a;
#pragma unroll
      for (int i = 0; i < 8; ++i) { a[i] = lo[i]; a[i + 8] = hi[i]; }
      acc = __builtin_amdgcn_wmma_f32_16x16x32_f16(
          false, a, false, bfrag[kc], (short)0, acc, false, false);
    }
    // C/D layout: VGPR v -> row M=v(+8 for hi lanes), lane -> column
#pragma unroll
    for (int v = 0; v < 8; ++v) {
      const int M = v + (hi16 ? 8 : 0);
      const int j = j0 + Ml;
      float aNew = mArr[M] + __logf(acc[v]) + emitS[M][j];
      float mk   = mkArr[M];
      float aOld = alphaS[M][j];
      alphaS[M][j] = aNew * mk + aOld * (1.0f - mk);
    }
    __syncthreads();
  }

  // ---- log_denominator = logsumexp_j(alpha + end) ----
  {
    float4 t0 = ((const float4*)(endT + c0))[0];
    float4 t1 = ((const float4*)(endT + c0))[1];
    float et[8] = {t0.x, t0.y, t0.z, t0.w, t1.x, t1.y, t1.z, t1.w};
    float av[8];
#pragma unroll
    for (int i = 0; i < 8; ++i) av[i] = alphaS[r][c0 + i] + et[i];
    float mx = av[0];
#pragma unroll
    for (int i = 1; i < 8; ++i) mx = fmaxf(mx, av[i]);
#pragma unroll
    for (int off = 1; off < 16; off <<= 1) mx = fmaxf(mx, __shfl_xor(mx, off));
    float s = 0.f;
#pragma unroll
    for (int i = 0; i < 8; ++i) s += __expf(av[i] - mx);
#pragma unroll
    for (int off = 1; off < 16; off <<= 1) s += __shfl_xor(s, off);
    if (q == 0) logden[b0 + r] = mx + __logf(s);
  }
}

// ---------------------------------------------------------------------------
// k2: gold-path score per sequence (label/transition/emission gathers)
// ---------------------------------------------------------------------------
__global__ __launch_bounds__(256) void crf_score(
    const float* __restrict__ in, const float* __restrict__ trans,
    const float* __restrict__ startT, const float* __restrict__ endT,
    const int* __restrict__ lab, const int* __restrict__ mask,
    float* __restrict__ scoreA, float* __restrict__ maskA) {
  __shared__ float sred[256];
  __shared__ float mred[256];
  const int b = blockIdx.x, tid = threadIdx.x;
  const int*   labB = lab  + (size_t)b * Lsz;
  const int*   mB   = mask + (size_t)b * Lsz;
  const float* inB  = in   + (size_t)b * Lsz * Tsz;
  float s = 0.f, ms = 0.f;
  for (int l = tid; l < Lsz; l += 256) {
    float mf = (float)mB[l];
    ms += mf;
    if (l >= 1)       s += trans[labB[l - 1] * Tsz + labB[l]] * mf;   // mask[:,1:]
    if (l <= Lsz - 2) s += inB[(size_t)l * Tsz + labB[l]] * mf;       // mask[:,:-1]
  }
  sred[tid] = s; mred[tid] = ms;
  __syncthreads();
  for (int st = 128; st > 0; st >>= 1) {
    if (tid < st) { sred[tid] += sred[tid + st]; mred[tid] += mred[tid + st]; }
    __syncthreads();
  }
  if (tid == 0) {
    float msum   = mred[0];
    int  lastIdx = (int)msum - 1;
    int  lastTag = labB[lastIdx];
    float sc = startT[labB[0]] + sred[0] + endT[lastTag] +
               inB[(size_t)(Lsz - 1) * Tsz + lastTag] * (float)mB[Lsz - 1];
    scoreA[b] = sc; maskA[b] = msum;
  }
}

// ---------------------------------------------------------------------------
// k3: deterministic tree reduce -> scalar NLL
// ---------------------------------------------------------------------------
__global__ __launch_bounds__(512) void crf_reduce(
    const float* __restrict__ logden, const float* __restrict__ scoreA,
    const float* __restrict__ maskA, float* __restrict__ out) {
  __shared__ float r1[512];
  __shared__ float r2[512];
  const int t = threadIdx.x;
  r1[t] = logden[t] - scoreA[t];
  r2[t] = maskA[t];
  __syncthreads();
  for (int s = 256; s > 0; s >>= 1) {
    if (t < s) { r1[t] += r1[t + s]; r2[t] += r2[t + s]; }
    __syncthreads();
  }
  if (t == 0) out[0] = r1[0] / r2[0];
}

extern "C" void kernel_launch(void* const* d_in, const int* in_sizes, int n_in,
                              void* d_out, int out_size, void* d_ws, size_t ws_size,
                              hipStream_t stream) {
  const float* in     = (const float*)d_in[0];
  const float* trans  = (const float*)d_in[1];
  const float* startT = (const float*)d_in[2];
  const float* endT   = (const float*)d_in[3];
  const int*   lab    = (const int*)d_in[4];
  const int*   mask   = (const int*)d_in[5];
  float* out = (float*)d_out;

  char* ws = (char*)d_ws;
  _Float16* expT  = (_Float16*)ws;                              // 32 KB
  float*    logden = (float*)(ws + (size_t)Tsz * Tsz * sizeof(_Float16));
  float*    scoreA = logden + Bsz;
  float*    maskA  = scoreA + Bsz;

  crf_prep_expT<<<(Tsz * Tsz + 255) / 256, 256, 0, stream>>>(trans, expT);
  crf_forward<<<Bsz / BPG, 256, 0, stream>>>(in, startT, endT, mask, expT, logden);
  crf_score<<<Bsz, 256, 0, stream>>>(in, trans, startT, endT, lab, mask, scoreA, maskA);
  crf_reduce<<<1, 512, 0, stream>>>(logden, scoreA, maskA, out);
}
